// GraphUnetVarAdj_21010980012139
// MI455X (gfx1250) — compile-verified, matrix operand order checked by reading, not run on hardware
//
#include <hip/hip_runtime.h>
#include <hip/hip_bf16.h>
#include <math.h>

typedef __attribute__((ext_vector_type(2))) float v2f;
typedef __attribute__((ext_vector_type(8))) float v8f;
typedef __attribute__((ext_vector_type(2))) int   v2i;
typedef __attribute__((ext_vector_type(8))) int   v8i;

#define BATCH 8
#define D     128
#define N0    1024
#define K1    921
#define K1P   928
#define K1R   960   // round64(K1P): padded operand rows for iu8 GEMM strips
#define K2    644
#define K2P   656
#define K2R   704
#define K3    386
#define K3P   400
#define K3R   448
#define NSORT 1024

// ---------------------------------------------------------------------------
// T[b] (rowsP x 128) = A[b] (rowsP x colsP) @ X[b] (colsP x 128), fp32 WMMA.
// Register-blocked: one wave computes a 16x64 strip (4 N-tiles), loading each
// A fragment once per K-step and issuing 4 back-to-back WMMAs.
// grid = (rowsP/16, 2, BATCH), block = 32.
// ---------------------------------------------------------------------------
__global__ void adjx_wmma_f32(const float* __restrict__ A, const float* __restrict__ X,
                              float* __restrict__ T, int rowsP, int colsP) {
  const int lane = threadIdx.x;
  const int m  = lane & 15;
  const int hi = lane >> 4;                 // lanes 16..31 take K+2,K+3 / M+8
  const int n0 = blockIdx.y * 64;           // first column of this strip
  const float* Ab = A + (size_t)blockIdx.z * rowsP * colsP;
  const float* Xb = X + (size_t)blockIdx.z * colsP * D;
  float*       Tb = T + (size_t)blockIdx.z * rowsP * D;

  const float* ar = Ab + (size_t)(blockIdx.x * 16 + m) * colsP + hi * 2;
  const float* xc = Xb + n0 + m + (size_t)(hi * 2) * D;

  v8f acc0 = {}, acc1 = {}, acc2 = {}, acc3 = {};
  for (int k = 0; k < colsP; k += 4) {
    v2f a = *(const v2f*)(ar + k);          // A[M][k+2hi], A[M][k+2hi+1]
    const float* xr = xc + (size_t)k * D;   // X[k+2hi][n0 + 16j + m]
    v2f b0; b0.x = xr[0];  b0.y = xr[0 + D];
    v2f b1; b1.x = xr[16]; b1.y = xr[16 + D];
    v2f b2; b2.x = xr[32]; b2.y = xr[32 + D];
    v2f b3; b3.x = xr[48]; b3.y = xr[48 + D];
    acc0 = __builtin_amdgcn_wmma_f32_16x16x4_f32(false, a, false, b0, (short)0, acc0, false, false);
    acc1 = __builtin_amdgcn_wmma_f32_16x16x4_f32(false, a, false, b1, (short)0, acc1, false, false);
    acc2 = __builtin_amdgcn_wmma_f32_16x16x4_f32(false, a, false, b2, (short)0, acc2, false, false);
    acc3 = __builtin_amdgcn_wmma_f32_16x16x4_f32(false, a, false, b3, (short)0, acc3, false, false);
  }
#pragma unroll
  for (int v = 0; v < 8; ++v) {
    int r = blockIdx.x * 16 + v + hi * 8;   // C VGPR v: M=v (lanes<16) / v+8
    float* tr = Tb + (size_t)r * D + n0 + m;
    tr[0]  = acc0[v];
    tr[16] = acc1[v];
    tr[32] = acc2[v];
    tr[48] = acc3[v];
  }
}

// ---------------------------------------------------------------------------
// out = act(T @ W^T + bias) (+ resid), torch Linear convention W[o][d].
// 16x64 strip per wave. grid = (rowsP/16, 2, BATCH), block = 32.
// act: 0 = none, 1 = silu. resid may be nullptr. Stores rows < realRows.
// ---------------------------------------------------------------------------
__global__ void lin_wmma_f32(const float* __restrict__ T, const float* __restrict__ W,
                             const float* __restrict__ bias,
                             const float* __restrict__ resid, size_t residBatch,
                             float* __restrict__ out, size_t outBatch,
                             int rowsP, int realRows, int act) {
  const int lane = threadIdx.x;
  const int m  = lane & 15;
  const int hi = lane >> 4;
  const int n0 = blockIdx.y * 64;
  const float* Tb = T + (size_t)blockIdx.z * rowsP * D;
  const float* tr = Tb + (size_t)(blockIdx.x * 16 + m) * D + hi * 2;
  const float* w0 = W + (size_t)(n0 +  0 + m) * D + hi * 2;  // B[k][N] = W[N][k]
  const float* w1 = W + (size_t)(n0 + 16 + m) * D + hi * 2;
  const float* w2 = W + (size_t)(n0 + 32 + m) * D + hi * 2;
  const float* w3 = W + (size_t)(n0 + 48 + m) * D + hi * 2;

  v8f acc0 = {}, acc1 = {}, acc2 = {}, acc3 = {};
#pragma unroll 4
  for (int k = 0; k < D; k += 4) {
    v2f a  = *(const v2f*)(tr + k);
    v2f b0 = *(const v2f*)(w0 + k);
    v2f b1 = *(const v2f*)(w1 + k);
    v2f b2 = *(const v2f*)(w2 + k);
    v2f b3 = *(const v2f*)(w3 + k);
    acc0 = __builtin_amdgcn_wmma_f32_16x16x4_f32(false, a, false, b0, (short)0, acc0, false, false);
    acc1 = __builtin_amdgcn_wmma_f32_16x16x4_f32(false, a, false, b1, (short)0, acc1, false, false);
    acc2 = __builtin_amdgcn_wmma_f32_16x16x4_f32(false, a, false, b2, (short)0, acc2, false, false);
    acc3 = __builtin_amdgcn_wmma_f32_16x16x4_f32(false, a, false, b3, (short)0, acc3, false, false);
  }
  float*       ob = out + (size_t)blockIdx.z * outBatch;
  const float* rb = resid ? resid + (size_t)blockIdx.z * residBatch : nullptr;
  float accv[4];
#pragma unroll
  for (int v = 0; v < 8; ++v) {
    int r = blockIdx.x * 16 + v + hi * 8;
    if (r < realRows) {
      accv[0] = acc0[v]; accv[1] = acc1[v]; accv[2] = acc2[v]; accv[3] = acc3[v];
#pragma unroll
      for (int j = 0; j < 4; ++j) {
        int c = n0 + j * 16 + m;
        float x = accv[j] + bias[c];
        if (act) { float s = 1.f / (1.f + __expf(-x)); x *= s; }
        if (rb) x += rb[(size_t)r * D + c];
        ob[(size_t)r * D + c] = x;
      }
    }
  }
}

// ---------------------------------------------------------------------------
// scores[b,i] = sigmoid(h . pool_w + pool_b) for i < realRows, else -inf pad.
// ---------------------------------------------------------------------------
__global__ void scores_sig(const float* __restrict__ H, const float* __restrict__ pw,
                           const float* __restrict__ pb, float* __restrict__ S,
                           int rowsP, int realRows) {
  int t = blockIdx.x * blockDim.x + threadIdx.x;
  if (t >= BATCH * NSORT) return;
  int b = t / NSORT, r = t % NSORT;
  float outv;
  if (r < realRows) {
    const float4* h = (const float4*)(H + ((size_t)b * rowsP + r) * D);
    const float4* w = (const float4*)pw;
    float acc = 0.f;
#pragma unroll
    for (int i = 0; i < D / 4; ++i) {
      float4 hv = h[i], wv = w[i];
      acc += hv.x * wv.x + hv.y * wv.y + hv.z * wv.z + hv.w * wv.w;
    }
    acc += pb[0];
    outv = 1.f / (1.f + __expf(-acc));
  } else {
    outv = -INFINITY;
  }
  S[(size_t)b * NSORT + r] = outv;
}

// ---------------------------------------------------------------------------
// Per-batch top-kk via LDS bitonic sort of 1024 keys.
// key = flip(score) << 32 | (0xFFFFFFFF - i)  (desc score, asc index on ties).
// ---------------------------------------------------------------------------
__global__ void topk_bitonic(const float* __restrict__ S, int* __restrict__ idx,
                             float* __restrict__ vals, int kk) {
  __shared__ unsigned long long key[NSORT];
  const int b = blockIdx.x;
  for (int i = threadIdx.x; i < NSORT; i += blockDim.x) {
    unsigned u = __float_as_uint(S[(size_t)b * NSORT + i]);
    u = (u & 0x80000000u) ? ~u : (u | 0x80000000u);       // monotone flip
    key[i] = ((unsigned long long)u << 32) | (unsigned)(0xFFFFFFFFu - (unsigned)i);
  }
  __syncthreads();
  for (int k = 2; k <= NSORT; k <<= 1) {
    for (int j = k >> 1; j > 0; j >>= 1) {
      for (int i = threadIdx.x; i < NSORT; i += blockDim.x) {
        int p = i ^ j;
        if (p > i) {
          unsigned long long a = key[i], c = key[p];
          bool desc = ((i & k) == 0);
          if (desc ? (a < c) : (a > c)) { key[i] = c; key[p] = a; }
        }
      }
      __syncthreads();
    }
  }
  for (int i = threadIdx.x; i < kk; i += blockDim.x) {
    unsigned long long kv = key[i];
    unsigned u = (unsigned)(kv >> 32);
    u = (u & 0x80000000u) ? (u ^ 0x80000000u) : ~u;       // unflip
    vals[(size_t)b * kk + i] = __uint_as_float(u);
    idx[(size_t)b * kk + i]  = (int)(0xFFFFFFFFu - (unsigned)(kv & 0xFFFFFFFFu));
  }
}

// ---------------------------------------------------------------------------
// new_h[b,i,:] = H[b, idx[b,i], :] * vals[b,i]  (zero pad rows up to kkP).
// grid = BATCH*kkP blocks of D threads.
// ---------------------------------------------------------------------------
__global__ void gather_scale(const float* __restrict__ H, size_t hBatch,
                             const int* __restrict__ idx, const float* __restrict__ vals,
                             float* __restrict__ Ho, int kk, int kkP) {
  int b = blockIdx.x / kkP, i = blockIdx.x % kkP, d = threadIdx.x;
  float v = 0.f;
  if (i < kk) {
    int s = idx[(size_t)b * kk + i];
    v = H[hBatch * b + (size_t)s * D + d] * vals[(size_t)b * kk + i];
  }
  Ho[((size_t)b * kkP + i) * D + d] = v;
}

// ---------------------------------------------------------------------------
// Build u8 operands for the 2-hop GEMM (kkR rows incl. strip padding, all
// pads written 0 every call):
//   A [kkR x nK]  : A[i,k]  = (G[idx_i, k] != 0)
//   Bt[kkR x nK]  : Bt[j,k] = (G[k, idx_j] != 0)   (B columns gathered)
// ---------------------------------------------------------------------------
__global__ void build_ab_u8(const float* __restrict__ G, const int* __restrict__ idx,
                            unsigned char* __restrict__ A, unsigned char* __restrict__ Bt,
                            int nP, int nK, int kk, int kkR) {
  size_t t = blockIdx.x * (size_t)blockDim.x + threadIdx.x;
  size_t perB = (size_t)kkR * (nK / 4);
  if (t >= (size_t)BATCH * perB) return;
  int b = (int)(t / perB);
  size_t rem = t % perB;
  int i  = (int)(rem / (nK / 4));
  int k4 = (int)(rem % (nK / 4)) * 4;
  unsigned av = 0, bv = 0;
  if (i < kk && k4 < nP) {
    const float* Gb = G + (size_t)b * nP * nP;
    int ii = idx[(size_t)b * kk + i];
    const float* gr = Gb + (size_t)ii * nP + k4;
#pragma unroll
    for (int c = 0; c < 4; ++c) {
      if (gr[c] != 0.f)                          av |= 1u << (8 * c);
      if (Gb[(size_t)(k4 + c) * nP + ii] != 0.f) bv |= 1u << (8 * c);
    }
  }
  size_t o = (size_t)b * perB + (size_t)i * (nK / 4) + (k4 >> 2);
  ((unsigned*)A)[o]  = av;
  ((unsigned*)Bt)[o] = bv;
}

// ---------------------------------------------------------------------------
// C[b] (kkP x kkP) int32 = A (kkP x nK) @ Bt^T via V_WMMA_I32_16X16X64_IU8.
// 16x64 strip per wave (4 col tiles, operand rows padded to kkR so all reads
// are in-bounds); stores guarded to col < kkP.
// grid = (kkP/16, kkR/64, BATCH), block = 32.
// ---------------------------------------------------------------------------
__global__ void twohop_iu8(const unsigned char* __restrict__ A,
                           const unsigned char* __restrict__ Bt,
                           int* __restrict__ C, int kkP, int kkR, int nK) {
  const int lane = threadIdx.x;
  const int m  = lane & 15;
  const int hi = lane >> 4;
  const int nK4 = nK >> 2;
  const unsigned* Ab = (const unsigned*)A  + (size_t)blockIdx.z * kkR * nK4;
  const unsigned* Bb = (const unsigned*)Bt + (size_t)blockIdx.z * kkR * nK4;
  const unsigned* ap  = Ab + (size_t)(blockIdx.x * 16 + m) * nK4;
  const unsigned* bp0 = Bb + (size_t)(blockIdx.y * 64 +  0 + m) * nK4;
  const unsigned* bp1 = Bb + (size_t)(blockIdx.y * 64 + 16 + m) * nK4;
  const unsigned* bp2 = Bb + (size_t)(blockIdx.y * 64 + 32 + m) * nK4;
  const unsigned* bp3 = Bb + (size_t)(blockIdx.y * 64 + 48 + m) * nK4;

  v8i acc0 = {}, acc1 = {}, acc2 = {}, acc3 = {};
  for (int k = 0; k < nK4; k += 16) {       // 64 u8 K-values per iteration
    int kd = k + hi * 2;                    // lanes>=16 take dword chunks +2
    v2i a0 = *(const v2i*)(ap + kd);
    v2i a1 = *(const v2i*)(ap + kd + 4);
    v2i a2 = *(const v2i*)(ap + kd + 8);
    v2i a3 = *(const v2i*)(ap + kd + 12);
    v8i a;
    a[0]=a0.x; a[1]=a0.y; a[2]=a1.x; a[3]=a1.y; a[4]=a2.x; a[5]=a2.y; a[6]=a3.x; a[7]=a3.y;
#pragma unroll
    for (int j = 0; j < 4; ++j) {
      const unsigned* bp = (j == 0) ? bp0 : (j == 1) ? bp1 : (j == 2) ? bp2 : bp3;
      v2i b0 = *(const v2i*)(bp + kd);
      v2i b1 = *(const v2i*)(bp + kd + 4);
      v2i b2 = *(const v2i*)(bp + kd + 8);
      v2i b3 = *(const v2i*)(bp + kd + 12);
      v8i b;
      b[0]=b0.x; b[1]=b0.y; b[2]=b1.x; b[3]=b1.y; b[4]=b2.x; b[5]=b2.y; b[6]=b3.x; b[7]=b3.y;
      if (j == 0) acc0 = __builtin_amdgcn_wmma_i32_16x16x64_iu8(false, a, false, b, acc0, false, false);
      if (j == 1) acc1 = __builtin_amdgcn_wmma_i32_16x16x64_iu8(false, a, false, b, acc1, false, false);
      if (j == 2) acc2 = __builtin_amdgcn_wmma_i32_16x16x64_iu8(false, a, false, b, acc2, false, false);
      if (j == 3) acc3 = __builtin_amdgcn_wmma_i32_16x16x64_iu8(false, a, false, b, acc3, false, false);
    }
  }
  int* Cb = C + (size_t)blockIdx.z * kkP * kkP;
#pragma unroll
  for (int v = 0; v < 8; ++v) {
    int r = blockIdx.x * 16 + v + hi * 8;
    int c0 = blockIdx.y * 64 + m;
    if (c0      < kkP) Cb[(size_t)r * kkP + c0]      = acc0[v];
    if (c0 + 16 < kkP) Cb[(size_t)r * kkP + c0 + 16] = acc1[v];
    if (c0 + 32 < kkP) Cb[(size_t)r * kkP + c0 + 32] = acc2[v];
    if (c0 + 48 < kkP) Cb[(size_t)r * kkP + c0 + 48] = acc3[v];
  }
}

// ---------------------------------------------------------------------------
// Row-normalize 2-hop counts: G[i,j] = (C[i,j]!=0) / deg_i. Pad rows -> 0.
// One 128-thread block per row.
// ---------------------------------------------------------------------------
__global__ void normg(const int* __restrict__ C, float* __restrict__ G, int kkP) {
  int b = blockIdx.x / kkP, i = blockIdx.x % kkP;
  const int* cr = C + ((size_t)b * kkP + i) * kkP;
  int cnt = 0;
  for (int j = threadIdx.x; j < kkP; j += blockDim.x) cnt += (cr[j] != 0) ? 1 : 0;
  __shared__ int red[128];
  red[threadIdx.x] = cnt;
  __syncthreads();
  for (int s = 64; s > 0; s >>= 1) {
    if ((int)threadIdx.x < s) red[threadIdx.x] += red[threadIdx.x + s];
    __syncthreads();
  }
  float inv = red[0] > 0 ? 1.f / (float)red[0] : 0.f;
  float* gr = G + ((size_t)b * kkP + i) * kkP;
  for (int j = threadIdx.x; j < kkP; j += blockDim.x) gr[j] = (cr[j] != 0) ? inv : 0.f;
}

__global__ void fill_zero(float* __restrict__ p, size_t n) {
  for (size_t i = blockIdx.x * (size_t)blockDim.x + threadIdx.x; i < n;
       i += (size_t)gridDim.x * blockDim.x)
    p[i] = 0.f;
}

// full[b, idx[b,i], :] = H[b, i, :]
__global__ void scatter_rows(const float* __restrict__ H, size_t hBatch,
                             const int* __restrict__ idx, float* __restrict__ Full,
                             int fullRowsP, int kk) {
  int b = blockIdx.x / kk, i = blockIdx.x % kk, d = threadIdx.x;
  int dst = idx[(size_t)b * kk + i];
  Full[((size_t)b * fullRowsP + dst) * D + d] = H[hBatch * b + (size_t)i * D + d];
}

__global__ void add_out(const float* __restrict__ a, const float* __restrict__ c,
                        float* __restrict__ o, size_t n) {
  for (size_t i = blockIdx.x * (size_t)blockDim.x + threadIdx.x; i < n;
       i += (size_t)gridDim.x * blockDim.x)
    o[i] = a[i] + c[i];
}

// ---------------------------------------------------------------------------
extern "C" void kernel_launch(void* const* d_in, const int* in_sizes, int n_in,
                              void* d_out, int out_size, void* d_ws, size_t ws_size,
                              hipStream_t stream) {
  (void)in_sizes; (void)n_in; (void)out_size; (void)ws_size;
  const float* g0   = (const float*)d_in[0];
  const float* hOrg = (const float*)d_in[1];
  const float* dW   = (const float*)d_in[2];
  const float* db   = (const float*)d_in[3];
  const float* uW   = (const float*)d_in[4];
  const float* ub   = (const float*)d_in[5];
  const float* pw   = (const float*)d_in[6];
  const float* pb   = (const float*)d_in[7];
  const float* bW   = (const float*)d_in[8];
  const float* bbp  = (const float*)d_in[9];
  float* out = (float*)d_out;

  char* wsp = (char*)d_ws;
  size_t off = 0;
  auto alloc = [&](size_t bytes) -> void* {
    void* p = wsp + off;
    off += (bytes + 255) & ~(size_t)255;
    return p;
  };

  float* g1 = (float*)alloc((size_t)BATCH * K1P * K1P * 4);
  float* g2 = (float*)alloc((size_t)BATCH * K2P * K2P * 4);
  float* g3 = (float*)alloc((size_t)BATCH * K3P * K3P * 4);
  float* d0 = (float*)alloc((size_t)BATCH * N0  * D * 4);
  float* d1 = (float*)alloc((size_t)BATCH * K1P * D * 4);
  float* d2 = (float*)alloc((size_t)BATCH * K2P * D * 4);
  float* Tt = (float*)alloc((size_t)BATCH * N0  * D * 4);
  float* h1 = (float*)alloc((size_t)BATCH * K1P * D * 4);
  float* h2 = (float*)alloc((size_t)BATCH * K2P * D * 4);
  float* h3 = (float*)alloc((size_t)BATCH * K3P * D * 4);
  float* hb = (float*)alloc((size_t)BATCH * K3P * D * 4);
  float* fullb = (float*)alloc((size_t)BATCH * N0 * D * 4);
  float* S  = (float*)alloc((size_t)BATCH * NSORT * 4);
  int*   idx0 = (int*)alloc((size_t)BATCH * K1 * 4);
  float* val0 = (float*)alloc((size_t)BATCH * K1 * 4);
  int*   idx1 = (int*)alloc((size_t)BATCH * K2 * 4);
  float* val1 = (float*)alloc((size_t)BATCH * K2 * 4);
  int*   idx2 = (int*)alloc((size_t)BATCH * K3 * 4);
  float* val2 = (float*)alloc((size_t)BATCH * K3 * 4);
  unsigned char* Au8 = (unsigned char*)alloc((size_t)BATCH * K1R * 1024);
  unsigned char* Bu8 = (unsigned char*)alloc((size_t)BATCH * K1R * 1024);
  int* C32 = (int*)alloc((size_t)BATCH * K1P * K1P * 4);

  const dim3 wv(32);

  // ---------------- Down level 0 (n = 1024) ----------------
  adjx_wmma_f32<<<dim3(N0/16, 2, BATCH), wv, 0, stream>>>(g0, hOrg, Tt, N0, N0);
  lin_wmma_f32<<<dim3(N0/16, 2, BATCH), wv, 0, stream>>>(
      Tt, dW + 0*D*D, db + 0*D, nullptr, 0, d0, (size_t)N0*D, N0, N0, 1);
  scores_sig<<<(BATCH*NSORT)/256, 256, 0, stream>>>(d0, pw + 0*D, pb + 0, S, N0, N0);
  topk_bitonic<<<BATCH, 256, 0, stream>>>(S, idx0, val0, K1);
  gather_scale<<<BATCH*K1P, D, 0, stream>>>(d0, (size_t)N0*D, idx0, val0, h1, K1, K1P);
  {
    size_t total = (size_t)BATCH * K1R * (1024/4);
    build_ab_u8<<<(unsigned)((total+255)/256), 256, 0, stream>>>(
        g0, idx0, Au8, Bu8, N0, 1024, K1, K1R);
  }
  twohop_iu8<<<dim3(K1P/16, K1R/64, BATCH), wv, 0, stream>>>(Au8, Bu8, C32, K1P, K1R, 1024);
  normg<<<BATCH*K1P, 128, 0, stream>>>(C32, g1, K1P);

  // ---------------- Down level 1 (n = 921 / 928) ----------------
  adjx_wmma_f32<<<dim3(K1P/16, 2, BATCH), wv, 0, stream>>>(g1, h1, Tt, K1P, K1P);
  lin_wmma_f32<<<dim3(K1P/16, 2, BATCH), wv, 0, stream>>>(
      Tt, dW + 1*D*D, db + 1*D, nullptr, 0, d1, (size_t)K1P*D, K1P, K1P, 1);
  scores_sig<<<(BATCH*NSORT)/256, 256, 0, stream>>>(d1, pw + 1*D, pb + 1, S, K1P, K1);
  topk_bitonic<<<BATCH, 256, 0, stream>>>(S, idx1, val1, K2);
  gather_scale<<<BATCH*K2P, D, 0, stream>>>(d1, (size_t)K1P*D, idx1, val1, h2, K2, K2P);
  {
    const int nK = 960;                                  // round64(928)
    size_t total = (size_t)BATCH * K2R * (nK/4);
    build_ab_u8<<<(unsigned)((total+255)/256), 256, 0, stream>>>(
        g1, idx1, Au8, Bu8, K1P, nK, K2, K2R);
    twohop_iu8<<<dim3(K2P/16, K2R/64, BATCH), wv, 0, stream>>>(Au8, Bu8, C32, K2P, K2R, nK);
  }
  normg<<<BATCH*K2P, 128, 0, stream>>>(C32, g2, K2P);

  // ---------------- Down level 2 (n = 644 / 656) ----------------
  adjx_wmma_f32<<<dim3(K2P/16, 2, BATCH), wv, 0, stream>>>(g2, h2, Tt, K2P, K2P);
  lin_wmma_f32<<<dim3(K2P/16, 2, BATCH), wv, 0, stream>>>(
      Tt, dW + 2*D*D, db + 2*D, nullptr, 0, d2, (size_t)K2P*D, K2P, K2P, 1);
  scores_sig<<<(BATCH*NSORT)/256, 256, 0, stream>>>(d2, pw + 2*D, pb + 2, S, K2P, K2);
  topk_bitonic<<<BATCH, 256, 0, stream>>>(S, idx2, val2, K3);
  gather_scale<<<BATCH*K3P, D, 0, stream>>>(d2, (size_t)K2P*D, idx2, val2, h3, K3, K3P);
  {
    const int nK = 704;                                  // round64(656)
    size_t total = (size_t)BATCH * K3R * (nK/4);
    build_ab_u8<<<(unsigned)((total+255)/256), 256, 0, stream>>>(
        g2, idx2, Au8, Bu8, K2P, nK, K3, K3R);
    twohop_iu8<<<dim3(K3P/16, K3R/64, BATCH), wv, 0, stream>>>(Au8, Bu8, C32, K3P, K3R, nK);
  }
  normg<<<BATCH*K3P, 128, 0, stream>>>(C32, g3, K3P);

  // ---------------- Bottleneck (n = 386 / 400) ----------------
  adjx_wmma_f32<<<dim3(K3P/16, 2, BATCH), wv, 0, stream>>>(g3, h3, Tt, K3P, K3P);
  lin_wmma_f32<<<dim3(K3P/16, 2, BATCH), wv, 0, stream>>>(
      Tt, bW, bbp, nullptr, 0, hb, (size_t)K3P*D, K3P, K3P, 0);

  float* out0 = out;                                // hs[0]: [B,644,128]
  float* out1 = out0 + (size_t)BATCH * K2 * D;      // hs[1]: [B,921,128]
  float* out2 = out1 + (size_t)BATCH * K1 * D;      // hs[2]: [B,1024,128]
  float* out3 = out2 + (size_t)BATCH * N0 * D;      // hs[3]: [B,1024,128]

  // ---------------- Up 0 (level 2) ----------------
  fill_zero<<<2048, 256, 0, stream>>>(fullb, (size_t)BATCH * K2P * D);
  scatter_rows<<<BATCH*K3, D, 0, stream>>>(hb, (size_t)K3P*D, idx2, fullb, K2P, K3);
  adjx_wmma_f32<<<dim3(K2P/16, 2, BATCH), wv, 0, stream>>>(g2, fullb, Tt, K2P, K2P);
  lin_wmma_f32<<<dim3(K2P/16, 2, BATCH), wv, 0, stream>>>(
      Tt, uW + 0*D*D, ub + 0*D, d2, (size_t)K2P*D, out0, (size_t)K2*D, K2P, K2, 1);

  // ---------------- Up 1 (level 1) ----------------
  fill_zero<<<2048, 256, 0, stream>>>(fullb, (size_t)BATCH * K1P * D);
  scatter_rows<<<BATCH*K2, D, 0, stream>>>(out0, (size_t)K2*D, idx1, fullb, K1P, K2);
  adjx_wmma_f32<<<dim3(K1P/16, 2, BATCH), wv, 0, stream>>>(g1, fullb, Tt, K1P, K1P);
  lin_wmma_f32<<<dim3(K1P/16, 2, BATCH), wv, 0, stream>>>(
      Tt, uW + 1*D*D, ub + 1*D, d1, (size_t)K1P*D, out1, (size_t)K1*D, K1P, K1, 1);

  // ---------------- Up 2 (level 0) ----------------
  fill_zero<<<2048, 256, 0, stream>>>(fullb, (size_t)BATCH * N0 * D);
  scatter_rows<<<BATCH*K1, D, 0, stream>>>(out1, (size_t)K1*D, idx0, fullb, N0, K1);
  adjx_wmma_f32<<<dim3(N0/16, 2, BATCH), wv, 0, stream>>>(g0, fullb, Tt, N0, N0);
  lin_wmma_f32<<<dim3(N0/16, 2, BATCH), wv, 0, stream>>>(
      Tt, uW + 2*D*D, ub + 2*D, d0, (size_t)N0*D, out2, (size_t)N0*D, N0, N0, 1);

  // hs[3] = hs[2] + org_h
  add_out<<<2048, 256, 0, stream>>>(out2, hOrg, out3, (size_t)BATCH * N0 * D);
}
